// LightweightHypersphericalAttention_25013889532213
// MI455X (gfx1250) — matrix operand order repaired and checked by reading, not compiled, not run on hardware
//
#include <hip/hip_runtime.h>
#include <hip/hip_bf16.h>
#include <math.h>

// ---------------- problem constants ----------------
#define B_   4
#define N_   2048
#define M_   2048
#define C_   512
#define H_   4
#define DV   128          // C/H
#define DQK  256          // 2C/H
#define SCALE_ 0.08838834764831845f   // 1/sqrt(128)

// padded LDS strides (in halves) to spread banks; multiples of 16 halves (32B)
#define KSTR 272          // 256 + 16 halves  -> 544B row stride
#define VSTR 48           // 32 + 16 halves   -> 96B row stride

typedef __attribute__((ext_vector_type(16))) _Float16 v16h;
typedef __attribute__((ext_vector_type(8)))  _Float16 v8h;
typedef __attribute__((ext_vector_type(8)))  float    v8f;

#define WMMA_F16(a,b,c) __builtin_amdgcn_wmma_f32_16x16x32_f16( \
    false, (a), false, (b), (short)0, (c), false, false)

__device__ __forceinline__ v8f vzero8() {
  v8f z;
#pragma unroll
  for (int i = 0; i < 8; ++i) z[i] = 0.0f;
  return z;
}

// issue one per-lane async copy: 16 bytes global -> LDS
__device__ __forceinline__ void async_cp16(unsigned lds_off, const void* gaddr) {
  asm volatile("global_load_async_to_lds_b128 %0, %1, off"
               :: "v"(lds_off), "v"((unsigned long long)(size_t)gaddr)
               : "memory");
}
__device__ __forceinline__ void wait_async0() {
  asm volatile("s_wait_asynccnt 0x0" ::: "memory");
}

// ---------------- kernel 1: f32 -> f16 convert ----------------
__global__ __launch_bounds__(256) void cvt_f16_kernel(
    const float* __restrict__ in, _Float16* __restrict__ out, int n) {
  int i = blockIdx.x * 256 + threadIdx.x;
  if (i < n) out[i] = (_Float16)in[i];
}

// ---------------- kernel 2: V transpose via LDS, vT[b,h][dv][M] -------------
// block: 32 m x 128 dv tile for one (b,h); coalesced reads and writes.
__global__ __launch_bounds__(256) void vT_kernel(
    const float* __restrict__ ctx, _Float16* __restrict__ vT) {
  __shared__ _Float16 tls[32 * 136];          // dv stride padded 128->136
  const int bh = blockIdx.x >> 6;             // 64 m-tiles per (b,h)
  const int m0 = (blockIdx.x & 63) * 32;
  const int b = bh >> 2, h = bh & (H_ - 1);
  const int tid = threadIdx.x;

#pragma unroll
  for (int p = 0; p < 16; ++p) {              // 4096 elems / 256 threads
    int lin = p * 256 + tid;
    int m = lin >> 7, dv = lin & 127;         // adjacent tid -> adjacent dv (coalesced read)
    tls[m * 136 + dv] =
        (_Float16)ctx[((size_t)(b * M_ + m0 + m)) * C_ + h * DV + dv];
  }
  __syncthreads();
#pragma unroll
  for (int p = 0; p < 16; ++p) {
    int lin = p * 256 + tid;
    int dv = lin >> 5, m = lin & 31;          // adjacent tid -> adjacent m (coalesced write)
    vT[((size_t)(bh * DV + dv)) * M_ + m0 + m] = tls[m * 136 + dv];
  }
}

// ---------------- kernel 3: fused QKV projection + per-head L2 norm ----------
__global__ __launch_bounds__(256) void qkv_norm_kernel(
    const float* __restrict__ x, const float* __restrict__ ctx,
    const _Float16* __restrict__ Wq, const float* __restrict__ radius,
    _Float16* __restrict__ qo, _Float16* __restrict__ ko) {
  __shared__ float ss[16 * H_];
  const int lane = threadIdx.x & 31;
  const int wave = threadIdx.x >> 5;
  const int half = lane >> 4;
  const int l16  = lane & 15;

  const bool isQ = blockIdx.x < 512;
  const int  r0  = (isQ ? blockIdx.x : blockIdx.x - 512) * 16;
  const float*    src = isQ ? x : ctx;
  _Float16*       dst = isQ ? qo : ko;

  if (threadIdx.x < 16 * H_) ss[threadIdx.x] = 0.0f;
  __syncthreads();

  v8f acc[8];
#pragma unroll
  for (int t = 0; t < 8; ++t) acc[t] = vzero8();

  const int k0 = half * 8;
  for (int kk = 0; kk < C_; kk += 32) {
    const float* ap = src + (size_t)(r0 + l16) * C_ + kk;
    v16h a;
#pragma unroll
    for (int j = 0; j < 8; ++j) {
      a[j]     = (_Float16)ap[2 * k0 + j];
      a[8 + j] = (_Float16)ap[2 * k0 + 8 + j];
    }
    // preload all 8 B fragments, then issue 8 WMMAs (staggered loadcnt waits)
    v16h bf[8];
#pragma unroll
    for (int t = 0; t < 8; ++t) {
      int o = wave * 128 + t * 16 + l16;
      bf[t] = *(const v16h*)(Wq + (size_t)o * C_ + kk + half * 16);
    }
#pragma unroll
    for (int t = 0; t < 8; ++t) acc[t] = WMMA_F16(a, bf[t], acc[t]);
  }

  const int head = wave >> 1;
#pragma unroll
  for (int i = 0; i < 8; ++i) {
    int nl = i + half * 8;
    float s = 0.0f;
#pragma unroll
    for (int t = 0; t < 8; ++t) { float v = acc[t][i]; s += v * v; }
    atomicAdd(&ss[nl * H_ + head], s);
  }
  __syncthreads();

  const float r_h = radius[head];
#pragma unroll
  for (int i = 0; i < 8; ++i) {
    int nl = i + half * 8;
    float nrm = sqrtf(ss[nl * H_ + head]);
    float sc  = r_h / fmaxf(nrm, 1e-12f);
    int rg = r0 + nl;
    int b  = rg >> 11;
    int n  = rg & 2047;
#pragma unroll
    for (int t = 0; t < 8; ++t) {
      int o = wave * 128 + t * 16 + l16;
      int d = o & (DQK - 1);
      dst[(((size_t)(b * H_ + head)) * N_ + n) * DQK + d] =
          (_Float16)(acc[t][i] * sc);
    }
  }
}

// ---------------- kernel 4: flash attention with async LDS staging ----------
// Block = 4 waves = 64 q rows of one (b,h). K/V m-block tiles are staged in
// LDS with global_load_async_to_lds_b128, double-buffered (ASYNCcnt).
__global__ __launch_bounds__(128) void flash_kernel(
    const _Float16* __restrict__ q, const _Float16* __restrict__ k,
    const _Float16* __restrict__ vT, _Float16* __restrict__ ao) {
  __shared__ _Float16 Kls[2][32 * KSTR];     // 2 x 17KB (padded rows)
  __shared__ _Float16 Vls[2][DV * VSTR];     // 2 x 12KB (padded rows)
  __shared__ _Float16 Pls[4][16 * 32];       // per-wave P transpose
  const int tid  = threadIdx.x;
  const int lane = tid & 31;
  const int wave = tid >> 5;
  const int half = lane >> 4;
  const int l16  = lane & 15;
  const int k0   = half * 8;

  const int bh = blockIdx.x >> 5;            // 32 blocks (64 rows each) per (b,h)
  const int nT4 = blockIdx.x & 31;
  const _Float16* qb = q  + ((size_t)bh * N_ + nT4 * 64 + wave * 16) * DQK;
  const _Float16* kbase = k  + (size_t)bh * M_ * DQK;
  const _Float16* vbase = vT + (size_t)bh * DV * M_;

  // resident q A-fragments (16 rows x 256 d)
  v16h qa[8];
#pragma unroll
  for (int kd = 0; kd < 8; ++kd) {
    const _Float16* p = qb + (size_t)l16 * DQK + kd * 32;
#pragma unroll
    for (int j = 0; j < 8; ++j) {
      qa[kd][j]     = p[2 * k0 + j];
      qa[kd][8 + j] = p[2 * k0 + 8 + j];
    }
  }

  // async stage of one m-block (32 k-rows, 32 v-cols) into LDS buffer `buf`
  auto issue_tile = [&](int buf, int m0) {
    unsigned kB = (unsigned)(size_t)&Kls[buf][0];
    const _Float16* gk = kbase + (size_t)m0 * DQK;
#pragma unroll
    for (int c = 0; c < 8; ++c) {            // 32 rows x 512B, row stride padded
      int lin = c * 128 + tid;               // 0..1023
      int row = lin >> 5, chunk = lin & 31;
      async_cp16(kB + (unsigned)(row * (KSTR * 2) + chunk * 16),
                 gk + (size_t)row * DQK + chunk * 8);
    }
    unsigned vB = (unsigned)(size_t)&Vls[buf][0];
    const _Float16* gv = vbase + (size_t)tid * M_ + m0;   // tid = dv row
#pragma unroll
    for (int c = 0; c < 4; ++c) {            // 128 rows x 64B
      async_cp16(vB + (unsigned)(tid * (VSTR * 2) + c * 16),
                 gv + c * 8);
    }
  };

  float m_run[8], l_run[8];
  v8f acc[8];
#pragma unroll
  for (int i = 0; i < 8; ++i) { m_run[i] = -INFINITY; l_run[i] = 0.0f; }
#pragma unroll
  for (int t = 0; t < 8; ++t) acc[t] = vzero8();

  issue_tile(0, 0);
  wait_async0();
  __syncthreads();

  for (int mb = 0; mb < M_ / 32; ++mb) {
    const int buf = mb & 1;
    if (mb + 1 < M_ / 32) issue_tile(buf ^ 1, (mb + 1) * 32);   // prefetch next
    const _Float16* Kt = &Kls[buf][0];
    const _Float16* Vt = &Vls[buf][0];

    // S = q . k^T, two 16x16 m-tiles, K=256 (B fragments from LDS)
    v8f s0 = vzero8(), s1 = vzero8();
#pragma unroll
    for (int kd = 0; kd < 8; ++kd) {
      v16h b0 = *(const v16h*)(Kt + (size_t)l16 * KSTR + kd * 32 + half * 16);
      s0 = WMMA_F16(qa[kd], b0, s0);
    }
#pragma unroll
    for (int kd = 0; kd < 8; ++kd) {
      v16h b1 = *(const v16h*)(Kt + (size_t)(16 + l16) * KSTR + kd * 32 + half * 16);
      s1 = WMMA_F16(qa[kd], b1, s1);
    }

    // online softmax (rows live across the 16 lanes of each half)
#pragma unroll
    for (int i = 0; i < 8; ++i) {
      float a0 = s0[i] * SCALE_, a1 = s1[i] * SCALE_;
      float mx = fmaxf(a0, a1);
      mx = fmaxf(mx, __shfl_xor(mx, 1, 32));
      mx = fmaxf(mx, __shfl_xor(mx, 2, 32));
      mx = fmaxf(mx, __shfl_xor(mx, 4, 32));
      mx = fmaxf(mx, __shfl_xor(mx, 8, 32));
      float mnew  = fmaxf(m_run[i], mx);
      float alpha = __expf(m_run[i] - mnew);
      float p0 = __expf(a0 - mnew);
      float p1 = __expf(a1 - mnew);
      float rs = p0 + p1;
      rs += __shfl_xor(rs, 1, 32);
      rs += __shfl_xor(rs, 2, 32);
      rs += __shfl_xor(rs, 4, 32);
      rs += __shfl_xor(rs, 8, 32);
      l_run[i] = l_run[i] * alpha + rs;
      m_run[i] = mnew;
      s0[i] = p0; s1[i] = p1;
#pragma unroll
      for (int t = 0; t < 8; ++t) acc[t][i] *= alpha;
    }

    // P: D-layout -> A-layout via LDS (same wave, in-order DS)
    _Float16* pw = &Pls[wave][0];
#pragma unroll
    for (int i = 0; i < 8; ++i) {
      int nl = i + half * 8;
      pw[nl * 32 + l16]      = (_Float16)s0[i];
      pw[nl * 32 + 16 + l16] = (_Float16)s1[i];
    }
    asm volatile("s_wait_dscnt 0x0" ::: "memory");
    v16h pa;
    {
      const _Float16* p = pw + l16 * 32;
#pragma unroll
      for (int j = 0; j < 8; ++j) {
        pa[j]     = p[2 * k0 + j];
        pa[8 + j] = p[2 * k0 + 8 + j];
      }
    }

    // out += P . V  (B fragments from LDS V tile)
#pragma unroll
    for (int t = 0; t < 8; ++t) {
      v16h bv = *(const v16h*)(Vt + (size_t)(t * 16 + l16) * VSTR + half * 16);
      acc[t] = WMMA_F16(pa, bv, acc[t]);
    }

    wait_async0();        // next-buffer transfers have landed
    __syncthreads();      // everyone done reading `buf`, writes of buf^1 visible
  }

  // epilogue: normalize by l, store f16 to ao[b][n][h*128+dv]
  const int b = bh >> 2, h = bh & (H_ - 1);
#pragma unroll
  for (int i = 0; i < 8; ++i) {
    float inv = 1.0f / fmaxf(l_run[i], 1e-20f);
    int n = (blockIdx.x & 31) * 64 + wave * 16 + i + half * 8;
#pragma unroll
    for (int t = 0; t < 8; ++t) {
      int col = h * DV + t * 16 + l16;
      ao[((size_t)(b * N_ + n)) * C_ + col] = (_Float16)(acc[t][i] * inv);
    }
  }
}

// ---------------- kernel 5: output projection ----------------
__global__ __launch_bounds__(256) void proj_kernel(
    const _Float16* __restrict__ ao, const _Float16* __restrict__ Wp,
    float* __restrict__ out) {
  const int lane = threadIdx.x & 31;
  const int wave = threadIdx.x >> 5;
  const int half = lane >> 4;
  const int l16  = lane & 15;
  const int k0   = half * 8;
  const int r0    = blockIdx.x * 16;
  const int obase = blockIdx.y * 128 + wave * 16;

  v8f acc = vzero8();
  for (int kk = 0; kk < C_; kk += 32) {
    const _Float16* ap = ao + (size_t)(r0 + l16) * C_ + kk;
    v16h a;
#pragma unroll
    for (int j = 0; j < 8; ++j) {
      a[j]     = ap[2 * k0 + j];
      a[8 + j] = ap[2 * k0 + 8 + j];
    }
    v16h bf = *(const v16h*)(Wp + (size_t)(obase + l16) * C_ + kk + half * 16);
    acc = WMMA_F16(a, bf, acc);
  }
#pragma unroll
  for (int i = 0; i < 8; ++i) {
    int r = r0 + i + half * 8;
    out[(size_t)r * C_ + obase + l16] = acc[i];
  }
}

// ---------------- launcher ----------------
extern "C" void kernel_launch(void* const* d_in, const int* in_sizes, int n_in,
                              void* d_out, int out_size, void* d_ws, size_t ws_size,
                              hipStream_t stream) {
  const float* x      = (const float*)d_in[0];
  const float* ctx    = (const float*)d_in[1];
  const float* Wqkv   = (const float*)d_in[2];
  const float* Wproj  = (const float*)d_in[3];
  const float* radius = (const float*)d_in[4];
  float* out = (float*)d_out;

  char* ws = (char*)d_ws;
  _Float16* Wq16 = (_Float16*)(ws + 0);          //  1 MB  (1024x512)
  _Float16* Wp16 = (_Float16*)(ws + 1048576);    //  0.5 MB (512x512)
  _Float16* q16  = (_Float16*)(ws + 1572864);    // 16 MB  [b,h][N][256]
  _Float16* k16  = (_Float16*)(ws + 18350080);   // 16 MB  [b,h][M][256]
  _Float16* vT16 = (_Float16*)(ws + 35127296);   //  8 MB  [b,h][dv][M]
  _Float16* ao16 = (_Float16*)(ws + 43515904);   //  8 MB  [b][n][C]

  cvt_f16_kernel<<<(1024 * 512) / 256, 256, 0, stream>>>(Wqkv, Wq16, 1024 * 512);
  cvt_f16_kernel<<<(512 * 512) / 256, 256, 0, stream>>>(Wproj, Wp16, 512 * 512);
  vT_kernel<<<B_ * H_ * (M_ / 32), 256, 0, stream>>>(ctx, vT16);
  qkv_norm_kernel<<<1024, 256, 0, stream>>>(x, ctx, Wq16, radius, q16, k16);
  flash_kernel<<<512, 128, 0, stream>>>(q16, k16, vT16, ao16);
  proj_kernel<<<dim3(512, 4), 256, 0, stream>>>(ao16, Wp16, out);
}
